// ResGCN_57767310131728
// MI455X (gfx1250) — matrix operand order repaired and checked by reading, not compiled, not run on hardware
//
#include <hip/hip_runtime.h>

typedef _Float16 v16h __attribute__((ext_vector_type(16)));
typedef _Float16 v8h  __attribute__((ext_vector_type(8)));
typedef float    v8f  __attribute__((ext_vector_type(8)));

#define NB   64
#define VV   17
#define TT0  300
#define M300 (NB * TT0 * VV)   // 326400

// ------------------------------------------------------------------ helpers
__device__ __forceinline__ v8h zero_v8h() {
  v8h z;
#pragma unroll
  for (int e = 0; e < 8; ++e) z[e] = (_Float16)0.f;
  return z;
}

// CDNA5 async global->LDS copy (16B per lane, tracked by ASYNCcnt).
// vdst VGPR holds the LDS byte address (= low 32 bits of the generic pointer,
// per the ISA aperture rule).
__device__ __forceinline__ void async_cp16(void* ldsDst, const void* gSrc) {
  unsigned lo = (unsigned)(unsigned long long)ldsDst;
  asm volatile("global_load_async_to_lds_b128 %0, %1, off"
               :: "v"(lo), "v"(gSrc) : "memory");
}
__device__ __forceinline__ void wait_async0() {
  asm volatile("s_wait_asynccnt 0x0" ::: "memory");
}

// WMMA tile engine.  LDS layout: lW = [128 o-rows][32 k + pad8] row-major,
// lX = [128 m-rows][32 k + pad8] row-major (per-lane fragment reads are
// contiguous for BOTH operands with the NHWC activation layout).
// 8 waves arranged 4(M=o) x 2(N=m); each wave computes 32(o) x 64(m).
__device__ __forceinline__ void wmma_tile_128(const _Float16* lW, const _Float16* lX,
                                              int lane, int wm, int wn, v8f (*acc)[4])
{
  v16h afr[2];
  v16h bfr[4];
#pragma unroll
  for (int im = 0; im < 2; ++im) {
    int row = wm * 32 + im * 16 + (lane & 15);
    int ko  = (lane >> 4) * 8;              // A frag: K 0..7/16..23 | 8..15/24..31
    v8h a0 = *(const v8h*)&lW[row * 40 + ko];
    v8h a1 = *(const v8h*)&lW[row * 40 + 16 + ko];
    v16h af;
#pragma unroll
    for (int e = 0; e < 8; ++e) { af[e] = a0[e]; af[e + 8] = a1[e]; }
    afr[im] = af;
  }
#pragma unroll
  for (int jn = 0; jn < 4; ++jn) {
    int row = wn * 64 + jn * 16 + (lane & 15);
    int ko  = (lane >> 4) * 8;              // B frag: K 0..7/16..23 | 8..15/24..31
    v8h b0 = *(const v8h*)&lX[row * 40 + ko];
    v8h b1 = *(const v8h*)&lX[row * 40 + 16 + ko];
    v16h bf;
#pragma unroll
    for (int e = 0; e < 8; ++e) { bf[e] = b0[e]; bf[e + 8] = b1[e]; }
    bfr[jn] = bf;
  }
#pragma unroll
  for (int im = 0; im < 2; ++im)
#pragma unroll
    for (int jn = 0; jn < 4; ++jn)
      acc[im][jn] = __builtin_amdgcn_wmma_f32_16x16x32_f16(
          false, afr[im], false, bfr[jn], (short)0, acc[im][jn], false, false);
}

// ------------------------------------------------------------------ GEMM (pointwise / residual)
// out[m*Mrows + o] = sum_c W[o*Kd + c] * X[icol(m)*ldX + c]
// icol maps output col (n*T_out+to)*V+v -> input row (n*T_in+to*stride)*V+v.
__global__ __launch_bounds__(256) void gemm_nt(const _Float16* __restrict__ W,
                                               const _Float16* __restrict__ X,
                                               float* __restrict__ O,
                                               int Mrows, int Kd, int NcolsOut, int ldX,
                                               int T_in, int T_out, int stride)
{
  __shared__ _Float16 lW[2][128 * 40];
  __shared__ _Float16 lX[2][128 * 40];
  const int tid  = threadIdx.x;
  const int lane = tid & 31;
  const int wave = tid >> 5;
  const int wm = wave & 3, wn = wave >> 2;
  const int nbb = blockIdx.x * 128;   // m tile
  const int mbb = blockIdx.y * 128;   // o tile

  // staging: each thread owns one row (srow) and half of its 32-k run (sel)
  const int srow = tid >> 1;
  const int sel  = (tid & 1) * 16;

  const int  grow = mbb + srow;
  const bool wv   = grow < Mrows;
  const _Float16* wp = W + (size_t)(wv ? grow : 0) * Kd + sel;

  const int  mo = nbb + srow;
  const bool xv = mo < NcolsOut;
  size_t icol = 0;
  if (xv) {
    if (stride == 1) icol = (size_t)mo;
    else {
      int nv = mo / VV; int v = mo - nv * VV;
      int nn = nv / T_out; int to = nv - nn * T_out;
      icol = (size_t)(nn * T_in + to * stride) * VV + v;
    }
  }
  const _Float16* xp = X + icol * ldX + sel;

  v8f acc[2][4];
#pragma unroll
  for (int i = 0; i < 2; ++i)
#pragma unroll
    for (int j = 0; j < 4; ++j)
#pragma unroll
      for (int e = 0; e < 8; ++e) acc[i][j][e] = 0.f;

  const int nk = Kd >> 5;

  auto stage = [&](int it, int buf) {
    int kb = it << 5;
    _Float16* dW = &lW[buf][srow * 40 + sel];
    _Float16* dX = &lX[buf][srow * 40 + sel];
    if (wv) {
      async_cp16(dW,     wp + kb);
      async_cp16(dW + 8, wp + kb + 8);
    } else {
      *(v8h*)dW = zero_v8h();
      *(v8h*)(dW + 8) = zero_v8h();
    }
    if (xv) {
      async_cp16(dX,     xp + kb);
      async_cp16(dX + 8, xp + kb + 8);
    } else {
      *(v8h*)dX = zero_v8h();
      *(v8h*)(dX + 8) = zero_v8h();
    }
  };

  stage(0, 0);
  for (int it = 0; it < nk; ++it) {
    const int buf = it & 1;
    wait_async0();
    __syncthreads();                       // buffer `buf` ready; buf^1 free
    if (it + 1 < nk) stage(it + 1, buf ^ 1);
    wmma_tile_128(lW[buf], lX[buf], lane, wm, wn, acc);
  }

  // store: per lane 8 consecutive o at fixed m -> vectorized b128 stores
#pragma unroll
  for (int im = 0; im < 2; ++im)
#pragma unroll
    for (int jn = 0; jn < 4; ++jn) {
      int m  = nbb + wn * 64 + jn * 16 + (lane & 15);
      int ob = mbb + wm * 32 + im * 16 + ((lane >> 4) << 3);
      if (m < NcolsOut && ob < Mrows)
        *(v8f*)(O + (size_t)m * Mrows + ob) = acc[im][jn];
    }
}

// ------------------------------------------------------------------ temporal conv (implicit GEMM)
// z[m][o] = sum_dt sum_ci Wt16[dt][o][ci] * Y16[(n*T_in + to*stride+dt-4)*V+v][ci]
// epilogue: relu(z + Res[m][o]) -> OutF/Out16 at row m, channel chOff+o.
__global__ __launch_bounds__(256) void tconv_nt(const _Float16* __restrict__ Wt16,
                                                const _Float16* __restrict__ Y16,
                                                const float* __restrict__ Res,
                                                float* __restrict__ OutF,
                                                _Float16* __restrict__ Out16,
                                                int Cout, int Cin, int NcolsOut,
                                                int T_in, int T_out, int stride,
                                                int ldOut, int chOff)
{
  __shared__ _Float16 lW[2][128 * 40];
  __shared__ _Float16 lX[2][128 * 40];
  const int tid  = threadIdx.x;
  const int lane = tid & 31;
  const int wave = tid >> 5;
  const int wm = wave & 3, wn = wave >> 2;
  const int nbb = blockIdx.x * 128;
  const int mbb = blockIdx.y * 128;

  const int srow = tid >> 1;
  const int sel  = (tid & 1) * 16;

  const int  grow = mbb + srow;
  const bool wv   = grow < Cout;

  const int  mo = nbb + srow;
  const bool cvv = mo < NcolsOut;
  int nn = 0, to = 0, vv = 0;
  if (cvv) { int nv = mo / VV; vv = mo - nv * VV; nn = nv / T_out; to = nv - nn * T_out; }

  const int nkc    = Cin >> 5;                  // 1,2,4,8
  const int kshift = (nkc == 1) ? 0 : (nkc == 2) ? 1 : (nkc == 4) ? 2 : 3;
  const int nit    = 9 * nkc;

  auto stage = [&](int it, int buf) {
    int dt = it >> kshift;
    int kb = (it & (nkc - 1)) << 5;
    _Float16* dW = &lW[buf][srow * 40 + sel];
    _Float16* dX = &lX[buf][srow * 40 + sel];
    if (wv) {
      const _Float16* wp = Wt16 + (size_t)dt * Cout * Cin
                                + (size_t)grow * Cin + kb + sel;
      async_cp16(dW,     wp);
      async_cp16(dW + 8, wp + 8);
    } else {
      *(v8h*)dW = zero_v8h();
      *(v8h*)(dW + 8) = zero_v8h();
    }
    int  ti   = to * stride + dt - 4;
    bool bval = cvv && (ti >= 0) && (ti < T_in);
    if (bval) {
      const _Float16* xp = Y16 + ((size_t)(nn * T_in + ti) * VV + vv) * Cin + kb + sel;
      async_cp16(dX,     xp);
      async_cp16(dX + 8, xp + 8);
    } else {
      *(v8h*)dX = zero_v8h();
      *(v8h*)(dX + 8) = zero_v8h();
    }
  };

  v8f acc[2][4];
#pragma unroll
  for (int i = 0; i < 2; ++i)
#pragma unroll
    for (int j = 0; j < 4; ++j)
#pragma unroll
      for (int e = 0; e < 8; ++e) acc[i][j][e] = 0.f;

  stage(0, 0);
  for (int it = 0; it < nit; ++it) {
    const int buf = it & 1;
    wait_async0();
    __syncthreads();                       // buffer `buf` ready; buf^1 free
    if (it + 1 < nit) stage(it + 1, buf ^ 1);
    wmma_tile_128(lW[buf], lX[buf], lane, wm, wn, acc);
  }

#pragma unroll
  for (int im = 0; im < 2; ++im)
#pragma unroll
    for (int jn = 0; jn < 4; ++jn) {
      int m  = nbb + wn * 64 + jn * 16 + (lane & 15);
      int ob = mbb + wm * 32 + im * 16 + ((lane >> 4) << 3);
      if (m < NcolsOut && ob < Cout) {
        v8f rv = *(const v8f*)(Res + (size_t)m * Cout + ob);
        v8f zv;
        v8h hv;
#pragma unroll
        for (int e = 0; e < 8; ++e) {
          float z = fmaxf(acc[im][jn][e] + rv[e], 0.f);
          zv[e] = z;
          hv[e] = (_Float16)z;
        }
        *(v8f*)(OutF  + (size_t)m * ldOut + chOff + ob) = zv;
        *(v8h*)(Out16 + (size_t)m * ldOut + chOff + ob) = hv;
      }
    }
}

// ------------------------------------------------------------------ adjacency mix + relu (VALU)
// Ymix16[(g*V+w)*cout + c] = relu( sum_k sum_v Ypre[(g*V+v)*KO + k*cout + c] * A[k][v][w] )
__global__ void adj_relu(const float* __restrict__ Ypre, const float* __restrict__ Aadj,
                         _Float16* __restrict__ Ymix16, int cout, int NTcnt)
{
  __shared__ float sA[3 * VV * VV];
  for (int i = threadIdx.x; i < 3 * VV * VV; i += blockDim.x) sA[i] = Aadj[i];
  __syncthreads();

  int idx = blockIdx.x * blockDim.x + threadIdx.x;
  int total = NTcnt * cout;
  if (idx >= total) return;
  int g = idx / cout;
  int c = idx - g * cout;
  int KO = 3 * cout;

  const float* base = Ypre + (size_t)g * VV * KO + c;
  float yv[3][VV];
#pragma unroll
  for (int k = 0; k < 3; ++k)
#pragma unroll
    for (int v = 0; v < VV; ++v) yv[k][v] = base[(size_t)v * KO + k * cout];

  _Float16* op = Ymix16 + (size_t)g * VV * cout + c;
#pragma unroll
  for (int w = 0; w < VV; ++w) {
    float s = 0.f;
#pragma unroll
    for (int k = 0; k < 3; ++k)
#pragma unroll
      for (int v = 0; v < VV; ++v) s += yv[k][v] * sA[(k * VV + v) * VV + w];
    op[(size_t)w * cout] = (_Float16)fmaxf(s, 0.f);
  }
}

// ------------------------------------------------------------------ conversions
__global__ void xcvt(const float* __restrict__ x, _Float16* __restrict__ h16, int branch)
{
  long long idx = (long long)blockIdx.x * blockDim.x + threadIdx.x;
  if (idx >= (long long)M300 * 32) return;
  int m = (int)(idx >> 5);
  int c = (int)(idx & 31);
  float f = 0.f;
  if (c < 6) {
    int n = m / (TT0 * VV);
    int r = m - n * (TT0 * VV);
    f = x[(((size_t)n * 3 + branch) * 6 + c) * (TT0 * VV) + r];
  }
  h16[idx] = (_Float16)f;
}

__global__ void wcvt_pad(const float* __restrict__ W, _Float16* __restrict__ W16,
                         int rows, int cin, int Cp)
{
  int idx = blockIdx.x * blockDim.x + threadIdx.x;
  if (idx >= rows * Cp) return;
  int r = idx / Cp, c = idx - r * Cp;
  W16[idx] = (_Float16)((c < cin) ? W[(size_t)r * cin + c] : 0.f);
}

__global__ void wtcvt(const float* __restrict__ Wt, _Float16* __restrict__ Wt16,
                      int cout, int cin)
{
  int idx = blockIdx.x * blockDim.x + threadIdx.x;
  if (idx >= 9 * cout * cin) return;
  int dt = idx / (cout * cin);
  int r  = idx - dt * cout * cin;
  int o  = r / cin, i = r - o * cin;
  Wt16[idx] = (_Float16)Wt[((size_t)o * cin + i) * 9 + dt];
}

// ------------------------------------------------------------------ head
__global__ void pool_mean(const float* __restrict__ H, float* __restrict__ pooled,
                          int C, int Mper)
{
  int idx = blockIdx.x * blockDim.x + threadIdx.x;
  if (idx >= NB * C) return;
  int n = idx / C, c = idx - n * C;
  const float* p = H + (size_t)n * Mper * C + c;
  float s = 0.f;
  for (int i = 0; i < Mper; ++i) s += p[(size_t)i * C];
  pooled[idx] = s / (float)Mper;
}

__global__ void fc_out1(const float* __restrict__ pooled, const float* __restrict__ W,
                        const float* __restrict__ b, float* __restrict__ finalb)
{
  int idx = blockIdx.x * blockDim.x + threadIdx.x;
  if (idx >= NB * 128) return;
  int n = idx / 128, j = idx - n * 128;
  float s = b[j];
  for (int c = 0; c < 128; ++c) s += pooled[n * 128 + c] * W[j * 128 + c];
  finalb[n * 256 + j] = s;
}

__global__ void max_t(const float* __restrict__ feat, float* __restrict__ xtemp)
{
  int idx = blockIdx.x * blockDim.x + threadIdx.x;
  if (idx >= NB * 96 * VV) return;
  int n = idx / (96 * VV);
  int r = idx - n * (96 * VV);
  int c = r / VV, v = r - c * VV;
  float m = -3.4e38f;
  for (int t = 0; t < TT0; ++t)
    m = fmaxf(m, feat[((size_t)(n * TT0 + t) * VV + v) * 96 + c]);
  xtemp[idx] = m;
}

__global__ void angle_kernel(const float* __restrict__ xtemp, const float* __restrict__ cls_w,
                             const float* __restrict__ cls_b, float* __restrict__ probe_out,
                             int* __restrict__ angle)
{
  int n = blockIdx.x * blockDim.x + threadIdx.x;
  if (n >= NB) return;
  float xg[96];
  for (int c = 0; c < 96; ++c) {
    float s = 0.f;
    for (int v = 0; v < VV; ++v) s += xtemp[(n * 96 + c) * VV + v];
    xg[c] = s / (float)VV;
  }
  float best = -3.4e38f; int bi = 0;
  for (int j = 0; j < 11; ++j) {
    float p = cls_b[j];
    for (int c = 0; c < 96; ++c) p += xg[c] * cls_w[j * 96 + c];
    probe_out[n * 11 + j] = p;
    if (p > best) { best = p; bi = j; }
  }
  angle[n] = bi;
}

__global__ void gem_clip(const float* __restrict__ xtemp, const float* __restrict__ gp,
                         float* __restrict__ z)
{
  int idx = blockIdx.x * blockDim.x + threadIdx.x;
  if (idx >= NB * 96 * VV) return;
  float p = gp[0];
  float x = fmaxf(xtemp[idx], 1e-6f);
  z[idx] = powf(powf(x, p), 1.f / p);   // mimic reference GeM-with-singleton-mean numerics
}

__global__ void rt_mean(const float* __restrict__ z, const float* __restrict__ tv,
                        const int* __restrict__ angle, float* __restrict__ meanrt)
{
  int idx = blockIdx.x * blockDim.x + threadIdx.x;
  if (idx >= NB * 96) return;
  int n = idx / 96, d = idx - n * 96;
  int a = angle[n];
  float s = 0.f;
  for (int v = 0; v < VV; ++v) {
    const float* tvp = tv + (((size_t)a * VV + v) * 96) * 96 + d;  // stride 96 over c
    for (int c = 0; c < 96; ++c) s += z[(n * 96 + c) * VV + v] * tvp[(size_t)c * 96];
  }
  meanrt[idx] = s / (float)VV;
}

__global__ void fc_out2(const float* __restrict__ meanrt, const float* __restrict__ W,
                        const float* __restrict__ b, float* __restrict__ finalb)
{
  int idx = blockIdx.x * blockDim.x + threadIdx.x;
  if (idx >= NB * 128) return;
  int n = idx / 128, j = idx - n * 128;
  float s = b[j];
  for (int d = 0; d < 96; ++d) s += meanrt[n * 96 + d] * W[j * 96 + d];
  finalb[n * 256 + 128 + j] = s;
}

__global__ void l2norm(const float* __restrict__ finalb, float* __restrict__ out)
{
  int n = blockIdx.x * blockDim.x + threadIdx.x;
  if (n >= NB) return;
  float s = 0.f;
  for (int i = 0; i < 256; ++i) { float v = finalb[n * 256 + i]; s += v * v; }
  float inv = 1.f / fmaxf(sqrtf(s), 1e-12f);
  for (int i = 0; i < 256; ++i) out[n * 256 + i] = finalb[n * 256 + i] * inv;
}

// ------------------------------------------------------------------ host orchestration
struct LayerCfg { int cin, cout, stride; };
static const LayerCfg BRCFG[3] = {{6,64,1},{64,64,1},{64,32,1}};
static const LayerCfg MNCFG[8] = {{96,128,2},{128,128,1},{128,128,1},{128,256,2},
                                  {256,256,1},{256,256,1},{256,128,1},{128,128,1}};

extern "C" void kernel_launch(void* const* d_in, const int* in_sizes, int n_in,
                              void* d_out, int out_size, void* d_ws, size_t ws_size,
                              hipStream_t stream)
{
  (void)n_in; (void)out_size; (void)ws_size;
  const float* x    = (const float*)d_in[0];
  const float* Aadj = (const float*)d_in[1];

  struct LP { const float *Wg, *Wt, *Wr; };
  LP lp[17];
  LayerCfg cfgs[17];
  for (int b = 0; b < 3; ++b)
    for (int l = 0; l < 3; ++l) cfgs[b * 3 + l] = BRCFG[l];
  for (int l = 0; l < 8; ++l) cfgs[9 + l] = MNCFG[l];

  int p = 2;
  for (int L = 0; L < 17; ++L) {
    LayerCfg c = cfgs[L];
    int wg_n = 3 * c.cout * c.cin;
    int wt_n = 9 * c.cout * c.cout;
    bool hasWr = !(c.cin == c.cout && c.stride == 1);
    int cnt = hasWr ? 3 : 2;
    lp[L].Wg = lp[L].Wt = lp[L].Wr = nullptr;
    for (int k = 0; k < cnt; ++k) {
      int sz = in_sizes[p];
      const float* ptr = (const float*)d_in[p];
      if (sz == wt_n)      lp[L].Wt = ptr;
      else if (sz == wg_n) lp[L].Wg = ptr;
      else                 lp[L].Wr = ptr;
      ++p;
    }
  }
  const float* gemp   = (const float*)d_in[p++];
  const float* transv = (const float*)d_in[p++];
  const float* cls_w  = (const float*)d_in[p++];
  const float* cls_b  = (const float*)d_in[p++];
  const float* fcn1_w = (const float*)d_in[p++];
  const float* fcn1_b = (const float*)d_in[p++];
  const float* fcn2_w = (const float*)d_in[p++];
  const float* fcn2_b = (const float*)d_in[p++];

  char* wsb = (char*)d_ws;
  size_t off = 0;
  auto carve = [&](size_t bytes) -> void* {
    void* r = wsb + off;
    off = (off + bytes + 255) & ~(size_t)255;
    return r;
  };
  const size_t PP = (size_t)64 * M300;
  _Float16* feat16 = (_Float16*)carve((size_t)96 * M300 * 2);   // [m][96]
  float*    featF  = (float*)   carve((size_t)96 * M300 * 4);   // [m][96]
  _Float16* hp16[2]; hp16[0] = (_Float16*)carve(PP * 2); hp16[1] = (_Float16*)carve(PP * 2);
  float*    hpF[2];  hpF[0]  = (float*)carve(PP * 4);    hpF[1]  = (float*)carve(PP * 4);
  float*    ypre   = (float*)   carve((size_t)384 * M300 * 4);  // [m][KO]
  _Float16* ymix16 = (_Float16*)carve((size_t)128 * M300 * 2);  // [m][cout]
  float*    resF   = (float*)   carve((size_t)64 * M300 * 4);   // [m][cout]
  _Float16* wg16   = (_Float16*)carve((size_t)768 * 256 * 2);
  _Float16* wt16   = (_Float16*)carve((size_t)9 * 256 * 256 * 2);
  _Float16* wr16   = (_Float16*)carve((size_t)256 * 256 * 2);
  float*    pooled = (float*)carve((size_t)NB * 128 * 4);
  float*    xtemp  = (float*)carve((size_t)NB * 96 * VV * 4);
  float*    zbuf   = (float*)carve((size_t)NB * 96 * VV * 4);
  float*    meanrt = (float*)carve((size_t)NB * 96 * 4);
  float*    finalb = (float*)carve((size_t)NB * 256 * 4);
  int*      angle  = (int*)carve((size_t)NB * 4);

  auto run_layer = [&](const LP& w, LayerCfg c, int T_in,
                       const _Float16* in16, const float* inF,
                       float* outF, _Float16* out16, int ldOut, int chOff) {
    int T_out = (c.stride == 2) ? T_in / 2 : T_in;
    int Min  = NB * T_in * VV;
    int Mout = NB * T_out * VV;
    int Cp   = (c.cin + 31) & ~31;
    int KO   = 3 * c.cout;

    wcvt_pad<<<(KO * Cp + 255) / 256, 256, 0, stream>>>(w.Wg, wg16, KO, c.cin, Cp);
    wtcvt<<<(9 * c.cout * c.cout + 255) / 256, 256, 0, stream>>>(w.Wt, wt16, c.cout, c.cout);

    // pointwise GEMM: ypre[m][KO]
    gemm_nt<<<dim3((Min + 127) / 128, (KO + 127) / 128), 256, 0, stream>>>(
        wg16, in16, ypre, KO, Cp, Min, Cp, T_in, T_in, 1);

    // adjacency mix + relu -> ymix16[m][cout]
    adj_relu<<<(NB * T_in * c.cout + 255) / 256, 256, 0, stream>>>(
        ypre, Aadj, ymix16, c.cout, NB * T_in);

    // residual
    const float* resPtr;
    if (w.Wr) {
      wcvt_pad<<<(c.cout * Cp + 255) / 256, 256, 0, stream>>>(w.Wr, wr16, c.cout, c.cin, Cp);
      gemm_nt<<<dim3((Mout + 127) / 128, (c.cout + 127) / 128), 256, 0, stream>>>(
          wr16, in16, resF, c.cout, Cp, Mout, Cp, T_in, T_out, c.stride);
      resPtr = resF;
    } else {
      resPtr = inF;  // identity: cin==cout, stride==1, ld == cout
    }

    // temporal conv (implicit GEMM) + residual + relu
    tconv_nt<<<dim3((Mout + 127) / 128, (c.cout + 127) / 128), 256, 0, stream>>>(
        wt16, ymix16, resPtr, outF, out16,
        c.cout, c.cout, Mout, T_in, T_out, c.stride, ldOut, chOff);
  };

  // ---- branches (outputs concat into [m][96] feature buffers)
  for (int b = 0; b < 3; ++b) {
    xcvt<<<(int)(((long long)M300 * 32 + 255) / 256), 256, 0, stream>>>(x, hp16[0], b);
    run_layer(lp[b * 3 + 0], BRCFG[0], TT0, hp16[0], nullptr, hpF[1], hp16[1], 64, 0);
    run_layer(lp[b * 3 + 1], BRCFG[1], TT0, hp16[1], hpF[1], hpF[0], hp16[0], 64, 0);
    run_layer(lp[b * 3 + 2], BRCFG[2], TT0, hp16[0], hpF[0], featF, feat16, 96, b * 32);
  }

  // ---- main stack
  const _Float16* cin16 = feat16;
  const float*    cinF  = featF;
  int T = TT0, cur = 0;
  for (int l = 0; l < 8; ++l) {
    LayerCfg c = MNCFG[l];
    run_layer(lp[9 + l], c, T, cin16, cinF, hpF[cur], hp16[cur], c.cout, 0);
    cin16 = hp16[cur]; cinF = hpF[cur];
    cur ^= 1;
    T = (c.stride == 2) ? T / 2 : T;
  }

  // ---- head
  float* out = (float*)d_out;
  pool_mean<<<(NB * 128 + 255) / 256, 256, 0, stream>>>(cinF, pooled, 128, T * VV);
  fc_out1<<<(NB * 128 + 255) / 256, 256, 0, stream>>>(pooled, fcn1_w, fcn1_b, finalb);
  max_t<<<(NB * 96 * VV + 255) / 256, 256, 0, stream>>>(featF, xtemp);
  angle_kernel<<<1, NB, 0, stream>>>(xtemp, cls_w, cls_b, out + NB * 256, angle);
  gem_clip<<<(NB * 96 * VV + 255) / 256, 256, 0, stream>>>(xtemp, gemp, zbuf);
  rt_mean<<<(NB * 96 + 255) / 256, 256, 0, stream>>>(zbuf, transv, angle, meanrt);
  fc_out2<<<(NB * 128 + 255) / 256, 256, 0, stream>>>(meanrt, fcn2_w, fcn2_b, finalb);
  l2norm<<<1, NB, 0, stream>>>(finalb, out);
}
